// BalancedMamba_36232344109505
// MI455X (gfx1250) — matrix operand order repaired and checked by reading, not compiled, not run on hardware
//
#include <hip/hip_runtime.h>
#include <hip/hip_bf16.h>

typedef __attribute__((ext_vector_type(16))) __bf16 v16bf;
typedef __attribute__((ext_vector_type(8)))  float  v8f;
typedef __attribute__((ext_vector_type(4)))  float  v4f;
typedef unsigned int u32x4 __attribute__((ext_vector_type(4)));
typedef int          i32x8 __attribute__((ext_vector_type(8)));
typedef int          i32x4 __attribute__((ext_vector_type(4)));

#define BSZ 16
#define LSEQ 4096
#define DIN 1280
#define DM 128
#define NST 16
#define RRK 8
#define NL 2
#define MTOT (BSZ * LSEQ)          // 65536 rows
#define PROJ_LD 48                 // dt(8) | B(16) | C(16) padded to 48

#if defined(__has_builtin)
#  if __has_builtin(__builtin_amdgcn_tensor_load_to_lds) && \
      __has_builtin(__builtin_amdgcn_s_wait_tensorcnt)
#    define HAVE_TDM 1
#  endif
#endif
#ifndef HAVE_TDM
#  define HAVE_TDM 0
#endif

// ---------------------------------------------------------------------------
// fp32 -> bf16 weight conversion
// ---------------------------------------------------------------------------
__global__ void f32_to_bf16_kernel(const float* __restrict__ in,
                                   __bf16* __restrict__ out, int n) {
  int i = blockIdx.x * blockDim.x + threadIdx.x;
  if (i < n) out[i] = (__bf16)in[i];
}

#if HAVE_TDM
// Issue a TDM load of a 2-D bf16 tile: 128 rows x 32 cols, row stride K
// elements, into LDS at lds_off_bytes (contiguous 128x32 slab).
// D# field packing per CDNA5 ISA.
// clang-23 arity: (u32x4 g0, i32x8 g1, i32x4 g2, i32x4 g3, i32x8, i32 cpol)
__device__ __forceinline__ void tdm_load_slab_bf16(
    const __bf16* gptr, unsigned lds_off_bytes, int K) {
  unsigned long long ga = (unsigned long long)(size_t)gptr;
  u32x4 g0;
  g0[0] = 1u;                                        // count=1, user mode
  g0[1] = lds_off_bytes;                             // lds_addr (bytes)
  g0[2] = (unsigned)(ga & 0xFFFFFFFFu);              // global_addr[31:0]
  g0[3] = (unsigned)((ga >> 32) & 0x01FFFFFFu)       // global_addr[56:32]
          | (2u << 30);                              // type = 2 ("image")
  i32x8 g1;
  g1[0] = (1 << 16);                                 // data_size=1 (2 bytes)
  g1[1] = (int)((unsigned)(K & 0xFFFF) << 16);       // tensor_dim0[15:0]
  g1[2] = (int)((((unsigned)K >> 16) & 0xFFFFu)      // tensor_dim0[31:16]
          | (128u << 16));                           // tensor_dim1[15:0]=128
  g1[3] = (32 << 16);                                // tile_dim0 = 32
  g1[4] = 128;                                       // tile_dim1=128, tile_dim2=0
  g1[5] = K;                                         // tensor_dim0_stride[31:0]
  g1[6] = 0;
  g1[7] = 0;
  i32x4 z4 = {0, 0, 0, 0};
  i32x8 z8 = {0, 0, 0, 0, 0, 0, 0, 0};
  __builtin_amdgcn_tensor_load_to_lds(g0, g1, z4, z4, z8, 0);
}
#endif

// ---------------------------------------------------------------------------
// LDS-staged GEMM: out[M x N] = A[M x K](fp32 -> bf16) * W[N x K]^T (bf16)
// Block = 8 waves; each wave owns a 16(M) x 128(N) tile (8 WMMAs / k-chunk).
// The 128x32 bf16 weight slab for each k-chunk is DMA'd into LDS by the
// Tensor Data Mover (double buffered, TENSORcnt), shared by all 8 waves.
// ---------------------------------------------------------------------------
__global__ __launch_bounds__(256) void gemm128_tdm_kernel(
    const float* __restrict__ A, const __bf16* __restrict__ W,
    const float* __restrict__ bias, const float* __restrict__ resid,
    float* __restrict__ out, int K, int ldout, int strips)
{
  const int wave = threadIdx.x >> 5;
  const int lane = threadIdx.x & 31;
  const int hh = lane >> 4;            // half-wave K-group
  const int ln = lane & 15;            // A row / B,C col within tile
  const int bm = blockIdx.x / strips;
  const int n0 = (blockIdx.x % strips) * 128;
  const int m0 = (bm * 8 + wave) * 16;
  const int NC = K >> 5;               // k-chunks of 32

  __shared__ alignas(128) __bf16 sB[2][128 * 32];   // 2 x 8KB slabs

  v8f c[8];
#pragma unroll
  for (int j = 0; j < 8; ++j) {
    if (resid) {
#pragma unroll
      for (int r = 0; r < 8; ++r)
        c[j][r] = resid[(size_t)(m0 + r + 8 * hh) * ldout + n0 + j * 16 + ln];
    } else {
      c[j] = (v8f){0.f, 0.f, 0.f, 0.f, 0.f, 0.f, 0.f, 0.f};
    }
  }

  const float* Arow = A + (size_t)(m0 + ln) * K;

#if HAVE_TDM
  if (wave == 0)
    tdm_load_slab_bf16(W + (size_t)n0 * K, (unsigned)(size_t)&sB[0][0], K);
#endif

  for (int kc = 0; kc < NC; ++kc) {
    const int cur = kc & 1;
    const int k0 = kc * 32;
#if HAVE_TDM
    if (wave == 0) __builtin_amdgcn_s_wait_tensorcnt(0);
    __syncthreads();                       // slab[cur] ready; slab[cur^1] free
    if (wave == 0 && kc + 1 < NC)
      tdm_load_slab_bf16(W + (size_t)n0 * K + (size_t)(kc + 1) * 32,
                         (unsigned)(size_t)&sB[cur ^ 1][0], K);
#else
    __syncthreads();
    for (int e = threadIdx.x; e < 128 * 32; e += 256) {
      int nn = e >> 5, kk = e & 31;
      sB[cur][e] = W[(size_t)(n0 + nn) * K + k0 + kk];
    }
    __syncthreads();
#endif
    // A fragment: K in {k0+hh*16 .. +15}
    const float* ap = Arow + k0 + hh * 16;
    __builtin_prefetch((const void*)(ap + 32), 0, 1);   // global_prefetch_b8
    v4f x0 = *(const v4f*)(ap);
    v4f x1 = *(const v4f*)(ap + 4);
    v4f x2 = *(const v4f*)(ap + 8);
    v4f x3 = *(const v4f*)(ap + 12);
    v16bf a;
#pragma unroll
    for (int i = 0; i < 4; ++i) {
      a[i]      = (__bf16)x0[i];
      a[4 + i]  = (__bf16)x1[i];
      a[8 + i]  = (__bf16)x2[i];
      a[12 + i] = (__bf16)x3[i];
    }
#pragma unroll
    for (int j = 0; j < 8; ++j) {
      const v16bf b =
          *(const v16bf*)(&sB[cur][(j * 16 + ln) * 32 + hh * 16]);
      c[j] = __builtin_amdgcn_wmma_f32_16x16x32_bf16(
          false, a, false, b, (short)0, c[j], false, false);
    }
  }

#pragma unroll
  for (int j = 0; j < 8; ++j) {
    const int n = n0 + j * 16 + ln;
    float bv = bias ? bias[n] : 0.f;
#pragma unroll
    for (int r = 0; r < 8; ++r)
      out[(size_t)(m0 + r + 8 * hh) * ldout + n] = c[j][r] + bv;
  }
}

// ---------------------------------------------------------------------------
// Small-N GEMM (x_proj, N=40 padded to 48): one 16x16 tile per wave.
// ---------------------------------------------------------------------------
__global__ __launch_bounds__(256) void gemm_small_wmma_kernel(
    const float* __restrict__ A, const __bf16* __restrict__ W,
    float* __restrict__ out, int N, int Nvalid, int K, int ldout)
{
  const int wave = threadIdx.x >> 5;
  const int lane = threadIdx.x & 31;
  const int hh = lane >> 4;
  const int ln = lane & 15;
  const int tilesN = (N + 15) >> 4;
  const int tn = wave;
  if (tn >= tilesN) return;
  const int m0 = blockIdx.x * 16;
  const int n  = tn * 16 + ln;

  v8f c = {0.f, 0.f, 0.f, 0.f, 0.f, 0.f, 0.f, 0.f};
  const float* Arow = A + (size_t)(m0 + ln) * K;

  for (int k0 = 0; k0 < K; k0 += 32) {
    const float* ap = Arow + k0 + hh * 16;
    v4f x0 = *(const v4f*)(ap);
    v4f x1 = *(const v4f*)(ap + 4);
    v4f x2 = *(const v4f*)(ap + 8);
    v4f x3 = *(const v4f*)(ap + 12);
    v16bf a;
#pragma unroll
    for (int i = 0; i < 4; ++i) {
      a[i]      = (__bf16)x0[i];
      a[4 + i]  = (__bf16)x1[i];
      a[8 + i]  = (__bf16)x2[i];
      a[12 + i] = (__bf16)x3[i];
    }
    v16bf b;
    if (n < Nvalid) {
      b = *(const v16bf*)(W + (size_t)n * K + k0 + hh * 16);
    } else {
#pragma unroll
      for (int i = 0; i < 16; ++i) b[i] = (__bf16)0.f;
    }
    c = __builtin_amdgcn_wmma_f32_16x16x32_bf16(
        false, a, false, b, (short)0, c, false, false);
  }

#pragma unroll
  for (int r = 0; r < 8; ++r) {
    int m = m0 + r + 8 * hh;
    if (n < Nvalid) out[(size_t)m * ldout + n] = c[r];
  }
}

// ---------------------------------------------------------------------------
// causal depthwise conv (k=2) + SiLU: su = silu(w0*u[t-1] + w1*u[t] + b)
// ---------------------------------------------------------------------------
__global__ void conv_silu_kernel(const float* __restrict__ xz,
                                 const float* __restrict__ cw,
                                 const float* __restrict__ cb,
                                 float* __restrict__ su) {
  int idx = blockIdx.x * blockDim.x + threadIdx.x;
  if (idx >= MTOT * DM) return;
  int d = idx & (DM - 1);
  size_t row = (size_t)(idx >> 7);          // b*L + t
  int t = (int)(row & (LSEQ - 1));
  float u  = xz[row * (2 * DM) + d];
  float up = (t > 0) ? xz[(row - 1) * (2 * DM) + d] : 0.f;
  float y = up * cw[d * 2 + 0] + u * cw[d * 2 + 1] + cb[d];
  su[idx] = y / (1.f + __expf(-y));
}

// ---------------------------------------------------------------------------
// delta = softplus(dt @ dt_w^T + dt_b), dt = proj[:, 0:8]
// ---------------------------------------------------------------------------
__global__ void dt_softplus_kernel(const float* __restrict__ proj,
                                   const float* __restrict__ dtw,
                                   const float* __restrict__ dtb,
                                   float* __restrict__ delta) {
  int idx = blockIdx.x * blockDim.x + threadIdx.x;
  if (idx >= MTOT * DM) return;
  int d = idx & (DM - 1);
  size_t m = (size_t)(idx >> 7);
  float s = dtb[d];
#pragma unroll
  for (int r = 0; r < RRK; ++r) s += proj[m * PROJ_LD + r] * dtw[d * RRK + r];
  delta[idx] = (s > 20.f) ? s : __logf(1.f + __expf(s));
}

// ---------------------------------------------------------------------------
// Selective scan + skip (u*Dp) + SiLU(z) gate; 16 states/thread in registers,
// B/C rows staged through LDS in chunks of 8 timesteps.
// ---------------------------------------------------------------------------
__global__ __launch_bounds__(128) void scan_kernel(
    const float* __restrict__ proj, const float* __restrict__ delta,
    const float* __restrict__ su, const float* __restrict__ xz,
    const float* __restrict__ A_log, const float* __restrict__ Dp,
    float* __restrict__ yout)
{
  const int b = blockIdx.x;
  const int d = threadIdx.x;
  float Ad[NST];
#pragma unroll
  for (int nn = 0; nn < NST; ++nn) Ad[nn] = -__expf(A_log[d * NST + nn]);
  const float Dd = Dp[d];
  float h[NST];
#pragma unroll
  for (int nn = 0; nn < NST; ++nn) h[nn] = 0.f;

  __shared__ float sBC[8 * 32];
  for (int tc = 0; tc < LSEQ; tc += 8) {
    __syncthreads();
#pragma unroll
    for (int i = 0; i < 2; ++i) {
      int e = threadIdx.x + i * 128;
      int row = e >> 5, col = e & 31;
      sBC[e] = proj[((size_t)b * LSEQ + tc + row) * PROJ_LD + 8 + col];
    }
    __syncthreads();
#pragma unroll
    for (int tt = 0; tt < 8; ++tt) {
      size_t m = (size_t)b * LSEQ + tc + tt;
      float dl = delta[m * DM + d];
      float uu = su[m * DM + d];
      float zz = xz[m * (2 * DM) + DM + d];
      float du = dl * uu;
      float y = 0.f;
#pragma unroll
      for (int nn = 0; nn < NST; ++nn) {
        float dA = __expf(dl * Ad[nn]);
        h[nn] = h[nn] * dA + du * sBC[tt * 32 + nn];
        y += h[nn] * sBC[tt * 32 + 16 + nn];
      }
      y += uu * Dd;
      float sz = zz / (1.f + __expf(-zz));
      yout[m * DM + d] = y * sz;
    }
  }
}

// ---------------------------------------------------------------------------
// LayerNorm over DM then mean-pool over L (accumulated via atomics)
// ---------------------------------------------------------------------------
__global__ void zero_kernel(float* __restrict__ p, int n) {
  int i = blockIdx.x * blockDim.x + threadIdx.x;
  if (i < n) p[i] = 0.f;
}

__global__ __launch_bounds__(128) void ln_pool_kernel(
    const float* __restrict__ hbuf, const float* __restrict__ g,
    const float* __restrict__ bb, float* __restrict__ p, int rowsPerBlock)
{
  const int chunks = LSEQ / 64;
  int b = blockIdx.x / chunks;
  int t0 = (blockIdx.x % chunks) * rowsPerBlock;
  int d = threadIdx.x;
  __shared__ float red[8];
  float acc = 0.f;
  for (int tt = 0; tt < rowsPerBlock; ++tt) {
    size_t m = (size_t)b * LSEQ + t0 + tt;
    float v = hbuf[m * DM + d];
    float s = v, s2 = v * v;
#pragma unroll
    for (int off = 16; off > 0; off >>= 1) {
      s  += __shfl_xor(s, off);
      s2 += __shfl_xor(s2, off);
    }
    int wv = d >> 5;
    if ((d & 31) == 0) { red[wv] = s; red[4 + wv] = s2; }
    __syncthreads();
    float S  = red[0] + red[1] + red[2] + red[3];
    float S2 = red[4] + red[5] + red[6] + red[7];
    __syncthreads();
    float mu  = S * (1.f / DM);
    float var = S2 * (1.f / DM) - mu * mu;
    acc += (v - mu) * rsqrtf(var + 1e-5f) * g[d] + bb[d];
  }
  atomicAdd(&p[b * DM + d], acc * (1.f / (float)LSEQ));
}

// ---------------------------------------------------------------------------
// Classifier: relu(p @ c1^T + b1) @ c2^T + b2  -> out[16,2]
// ---------------------------------------------------------------------------
__global__ __launch_bounds__(128) void classifier_kernel(
    const float* __restrict__ p, const float* __restrict__ c1w,
    const float* __restrict__ c1b, const float* __restrict__ c2w,
    const float* __restrict__ c2b, float* __restrict__ out)
{
  __shared__ float hid[16 * 64];
  int tid = threadIdx.x;
  for (int k = tid; k < 16 * 64; k += 128) {
    int b = k >> 6, j = k & 63;
    float s = c1b[j];
    for (int dd = 0; dd < DM; ++dd) s += p[b * DM + dd] * c1w[j * DM + dd];
    hid[k] = s > 0.f ? s : 0.f;
  }
  __syncthreads();
  if (tid < 32) {
    int b = tid >> 1, cc = tid & 1;
    float s = c2b[cc];
    for (int j = 0; j < 64; ++j) s += hid[b * 64 + j] * c2w[cc * 64 + j];
    out[b * 2 + cc] = s;
  }
}

// ---------------------------------------------------------------------------
extern "C" void kernel_launch(void* const* d_in, const int* in_sizes, int n_in,
                              void* d_out, int out_size, void* d_ws, size_t ws_size,
                              hipStream_t stream) {
  const float* x     = (const float*)d_in[0];
  const float* ip_w  = (const float*)d_in[1];
  const float* ip_b  = (const float*)d_in[2];
  const float* in_w  = (const float*)d_in[3];
  const float* cw    = (const float*)d_in[4];
  const float* cb    = (const float*)d_in[5];
  const float* xp_w  = (const float*)d_in[6];
  const float* dt_w  = (const float*)d_in[7];
  const float* dt_b  = (const float*)d_in[8];
  const float* A_log = (const float*)d_in[9];
  const float* Dp    = (const float*)d_in[10];
  const float* out_w = (const float*)d_in[11];
  const float* ln_g  = (const float*)d_in[12];
  const float* ln_b  = (const float*)d_in[13];
  const float* c1w   = (const float*)d_in[14];
  const float* c1b   = (const float*)d_in[15];
  const float* c2w   = (const float*)d_in[16];
  const float* c2b   = (const float*)d_in[17];

  char* ws = (char*)d_ws;
  size_t off = 0;
  auto alloc = [&](size_t bytes) -> void* {
    void* ptr = ws + off;
    off += (bytes + 255) & ~(size_t)255;
    return ptr;
  };
  float*  hbuf  = (float*) alloc((size_t)MTOT * DM * 4);
  float*  xz    = (float*) alloc((size_t)MTOT * 2 * DM * 4);
  float*  su    = (float*) alloc((size_t)MTOT * DM * 4);
  float*  proj  = (float*) alloc((size_t)MTOT * PROJ_LD * 4);
  float*  delta = (float*) alloc((size_t)MTOT * DM * 4);
  float*  yws   = (float*) alloc((size_t)MTOT * DM * 4);
  __bf16* ipwB  = (__bf16*)alloc((size_t)DM * DIN * 2);
  __bf16* inwB  = (__bf16*)alloc((size_t)NL * 2 * DM * DM * 2);
  __bf16* xpwB  = (__bf16*)alloc((size_t)NL * 40 * DM * 2);
  __bf16* outwB = (__bf16*)alloc((size_t)NL * DM * DM * 2);
  float*  pool  = (float*) alloc((size_t)BSZ * DM * 4);

  auto cvt = [&](const float* src, __bf16* dst, int n) {
    f32_to_bf16_kernel<<<(n + 255) / 256, 256, 0, stream>>>(src, dst, n);
  };
  cvt(ip_w,  ipwB,  DM * DIN);
  cvt(in_w,  inwB,  NL * 2 * DM * DM);
  cvt(xp_w,  xpwB,  NL * 40 * DM);
  cvt(out_w, outwB, NL * DM * DM);

  const int tilesM  = MTOT / 16;      // 4096
  const int blocksM = MTOT / 128;     // 512  (8 m-tiles per block)
  const int EW = MTOT * DM;

  // input_proj: h = x @ ip_w^T + ip_b   (N=128, K=1280)
  gemm128_tdm_kernel<<<blocksM * 1, 256, 0, stream>>>(
      x, ipwB, ip_b, nullptr, hbuf, DIN, DM, 1);

  for (int l = 0; l < NL; ++l) {
    // in_proj: xz = h @ in_w^T   (N=256 -> 2 strips of 128, K=128)
    gemm128_tdm_kernel<<<blocksM * 2, 256, 0, stream>>>(
        hbuf, inwB + (size_t)l * 2 * DM * DM, nullptr, nullptr, xz,
        DM, 2 * DM, 2);
    // conv + silu
    conv_silu_kernel<<<(EW + 255) / 256, 256, 0, stream>>>(
        xz, cw + l * DM * 2, cb + l * DM, su);
    // x_proj: proj = su @ xp_w^T   (N padded 40->48, K=128)
    gemm_small_wmma_kernel<<<tilesM, 256, 0, stream>>>(
        su, xpwB + (size_t)l * 40 * DM, proj, PROJ_LD, 40, DM, PROJ_LD);
    // dt_proj + softplus
    dt_softplus_kernel<<<(EW + 255) / 256, 256, 0, stream>>>(
        proj, dt_w + l * DM * RRK, dt_b + l * DM, delta);
    // selective scan + gate
    scan_kernel<<<BSZ, DM, 0, stream>>>(
        proj, delta, su, xz, A_log + l * DM * NST, Dp + l * DM, yws);
    // out_proj with fused residual: h = h + yws @ out_w^T
    gemm128_tdm_kernel<<<blocksM * 1, 256, 0, stream>>>(
        yws, outwB + (size_t)l * DM * DM, nullptr, hbuf, hbuf,
        DM, DM, 1);
  }

  zero_kernel<<<(BSZ * DM + 255) / 256, 256, 0, stream>>>(pool, BSZ * DM);
  ln_pool_kernel<<<BSZ * (LSEQ / 64), DM, 0, stream>>>(hbuf, ln_g, ln_b, pool, 64);
  classifier_kernel<<<1, 128, 0, stream>>>(pool, c1w, c1b, c2w, c2b,
                                           (float*)d_out);
}
